// STSelfAttention_72619307041440
// MI455X (gfx1250) — matrix operand order repaired and checked by reading, not compiled, not run on hardware
//
#include <hip/hip_runtime.h>
#include <hip/hip_bf16.h>

// ---------------------------------------------------------------------------
// STSelfAttention for MI455X (gfx1250, wave32, WMMA f16 path)
// B=16 T=12 N=325 D=64, HEAD_DIM=8
// ---------------------------------------------------------------------------

typedef _Float16 h16;
typedef __attribute__((ext_vector_type(16))) _Float16 v16h;
typedef __attribute__((ext_vector_type(8)))  _Float16 v8h;
typedef __attribute__((ext_vector_type(8)))  float    v8f;

#define BB   16
#define TT   12
#define NN   325
#define MROW (BB * TT * NN)   // 62400
#define ANP  336              // adp padded row stride (halfs, 16B-multiple)
#define SCALE_HD 0.35355339059327373f  // 8^-0.5
#define ASTRIDE 40   // padded LDS row stride (halfs) to spread bank groups
#define BSTRIDE 40

// A fragment (16-bit A, 16x32) is contiguous in K:
//   a[0..15] = K[16*halfsel .. +15]
__device__ __forceinline__ v16h load_a_frag(const h16* row, int halfsel) {
    union { v16h v; v8h p[2]; } u;
    u.p[0] = *(const v8h*)(row + 16 * halfsel);
    u.p[1] = *(const v8h*)(row + 16 * halfsel + 8);
    return u.v;
}
// B fragment b[i] = B[k=i+16*halfsel][col]; staged transposed [col][k].
__device__ __forceinline__ v16h load_b_fragT(const h16* rowT, int halfsel) {
    union { v16h v; v8h p[2]; } u;
    u.p[0] = *(const v8h*)(rowT + 16 * halfsel);
    u.p[1] = *(const v8h*)(rowT + 16 * halfsel + 8);
    return u.v;
}

// ---------------------------------------------------------------------------
// Weight prep: f16 [col][K] layouts (mostly straight conversions) + fused
// W_comb = W_r2 @ W_gcn and its bias.
// ---------------------------------------------------------------------------
__global__ void k_prep_weights(
    const float* __restrict__ W_r1, const float* __restrict__ W_gcn,
    const float* __restrict__ b_gcn, const float* __restrict__ W_r2,
    const float* __restrict__ b_r2,
    const float* __restrict__ Wtq, const float* __restrict__ Wtk,
    const float* __restrict__ Wtv,
    const float* __restrict__ Wgq, const float* __restrict__ Wgk,
    const float* __restrict__ Wgv,
    const float* __restrict__ W_proj,
    h16* __restrict__ Bw1, h16* __restrict__ Bcomb, float* __restrict__ biascomb,
    h16* __restrict__ Bt, h16* __restrict__ Bg, h16* __restrict__ Bproj)
{
    int gid = blockIdx.x * blockDim.x + threadIdx.x;
    int stride = gridDim.x * blockDim.x;
    // Bw1[c][k] = W_r1[c,k]  (32 x 64)
    for (int i = gid; i < 32 * 64; i += stride) Bw1[i] = (h16)W_r1[i];
    // Bcomb[d][c] = sum_o W_r2[d,o] * W_gcn[o,c]  (64 x 96)
    for (int i = gid; i < 64 * 96; i += stride) {
        int d = i / 96, c = i % 96;
        float s = 0.f;
        for (int o = 0; o < 32; ++o) s += W_r2[d * 32 + o] * W_gcn[o * 96 + c];
        Bcomb[i] = (h16)s;
    }
    for (int i = gid; i < 64; i += stride) {
        float s = b_r2[i];
        for (int o = 0; o < 32; ++o) s += W_r2[i * 32 + o] * b_gcn[o];
        biascomb[i] = s;
    }
    // Bt = [Wtq; Wtk; Wtv]  (48 x 64)
    for (int i = gid; i < 16 * 64; i += stride) {
        Bt[i]             = (h16)Wtq[i];
        Bt[16 * 64 + i]   = (h16)Wtk[i];
        Bt[32 * 64 + i]   = (h16)Wtv[i];
    }
    // Bg = [Wgq; Wgk; Wgv]  (96 x 64)
    for (int i = gid; i < 32 * 64; i += stride) {
        Bg[i]             = (h16)Wgq[i];
        Bg[32 * 64 + i]   = (h16)Wgk[i];
        Bg[64 * 64 + i]   = (h16)Wgv[i];
    }
    // Bproj[d][k] = W_proj[d,k]  (64 x 48)
    for (int i = gid; i < 64 * 48; i += stride) Bproj[i] = (h16)W_proj[i];
}

__global__ void k_f32_to_f16(const float* __restrict__ in, h16* __restrict__ out, long n) {
    long i = (long)blockIdx.x * blockDim.x + threadIdx.x;
    long stride = (long)gridDim.x * blockDim.x;
    for (; i < n; i += stride) out[i] = (h16)in[i];
}

// ---------------------------------------------------------------------------
// dgconstruct step 1: tmp1[b,j,k] = sum_i p1[ind[b]%288, i] * pk[i,j,k]
// ---------------------------------------------------------------------------
__global__ void k_tmp1(const float* __restrict__ p1, const int* __restrict__ ind,
                       const float* __restrict__ pk, float* __restrict__ tmp1)
{
    int b = blockIdx.x;
    int tid = threadIdx.x;
    __shared__ float te[40];
    if (tid < 40) {
        int day = ind[b] % 288;
        te[tid] = p1[(long)day * 40 + tid];
    }
    __syncthreads();
    for (int idx = tid; idx < 1600; idx += blockDim.x) {
        int j = idx / 40, k = idx % 40;
        float s = 0.f;
        for (int i = 0; i < 40; ++i) s += te[i] * pk[((long)i * 40 + j) * 40 + k];
        tmp1[(long)b * 1600 + idx] = s;
    }
}

// ---------------------------------------------------------------------------
// dgconstruct step 2+3: per (b,n): tmp2 = p2[n] @ tmp1[b]; raw = tmp2 @ p3^T;
// relu + softmax -> adp16[b,n,:] (rows padded to ANP=336 with zeros)
// ---------------------------------------------------------------------------
__global__ __launch_bounds__(128) void k_adp(
    const float* __restrict__ p2, const float* __restrict__ p3,
    const float* __restrict__ tmp1, h16* __restrict__ adp16)
{
    int bn = blockIdx.x;
    int b = bn / NN, n = bn % NN;
    int tid = threadIdx.x;
    __shared__ float t2[40];
    __shared__ float vals[336];
    __shared__ float part[128];

    if (tid < 40) {
        const float* t1 = tmp1 + (long)b * 1600;
        float s = 0.f;
        for (int j = 0; j < 40; ++j) s += p2[n * 40 + j] * t1[j * 40 + tid];
        t2[tid] = s;
    }
    __syncthreads();
    for (int m = tid; m < 336; m += 128) {
        float s = -1e30f;
        if (m < NN) {
            s = 0.f;
            for (int k = 0; k < 40; ++k) s += p3[m * 40 + k] * t2[k];
            s = s > 0.f ? s : 0.f;
        }
        vals[m] = s;
    }
    __syncthreads();
    float mx = -1e30f;
    for (int m = tid; m < NN; m += 128) mx = fmaxf(mx, vals[m]);
    part[tid] = mx;
    __syncthreads();
    for (int off = 64; off; off >>= 1) {
        if (tid < off) part[tid] = fmaxf(part[tid], part[tid + off]);
        __syncthreads();
    }
    mx = part[0];
    float sm = 0.f;
    for (int m = tid; m < NN; m += 128) {
        float e = __expf(vals[m] - mx);
        vals[m] = e;
        sm += e;
    }
    __syncthreads();
    part[tid] = sm;
    __syncthreads();
    for (int off = 64; off; off >>= 1) {
        if (tid < off) part[tid] += part[tid + off];
        __syncthreads();
    }
    float inv = 1.f / part[0];
    for (int m = tid; m < 336; m += 128)
        adp16[((long)b * NN + n) * ANP + m] = (m < NN) ? (h16)(vals[m] * inv)
                                                       : (h16)0.f;
}

// ---------------------------------------------------------------------------
// Row-GEMM (WMMA), templated on MODE so K/Ncols/out-type/row-remap are
// compile-time. A is always row-major-linear (stride K) -> chunked b128
// staging; B is [col][K] -> chunked b128 staging. Layout remaps happen only
// at C writeback (16 rows per wave).
// ---------------------------------------------------------------------------
#define MODE_H     0   // A=x16 (b,t,n)  K=64 N=32 -> hcat (b,n,t) ch 0..31
#define MODE_XG    1   // A=hcat (b,n,t) K=96 N=64 -> xg (b,t,n)
#define MODE_TQKV  2   // A=xg natural   K=64 N=48 -> tqkv natural, f32 out
#define MODE_GQKV  3   // A=xg natural   K=64 N=96 -> gqkv natural
#define MODE_PROJ  4   // A=att48        K=48 N=64 -> d_out, f32 out

template <int MODE>
__device__ __forceinline__ long c_off_row(int r) {
    if (MODE == MODE_H) {          // r = (b,t,n) -> hcat (b,n,t)
        int b = r / (TT * NN), rem = r % (TT * NN);
        int t = rem / NN, n = rem % NN;
        return ((long)(b * NN + n) * TT + t) * 96;
    }
    if (MODE == MODE_XG) {         // r = (b,n,t) -> xg (b,t,n)
        int b = r / (NN * TT), rem = r % (NN * TT);
        int n = rem / TT, t = rem % TT;
        return ((long)(b * TT + t) * NN + n) * 64;
    }
    if (MODE == MODE_TQKV) return (long)r * 48;
    if (MODE == MODE_GQKV) return (long)r * 96;
    return (long)r * 64;           // MODE_PROJ
}

template <int MODE>
__global__ __launch_bounds__(256) void k_row_gemm(
    const h16* __restrict__ A, const h16* __restrict__ Bwt,
    const float* __restrict__ bias, void* __restrict__ Cout, int M)
{
    const int K  = (MODE == MODE_XG) ? 96 : (MODE == MODE_PROJ) ? 48 : 64;
    const int NC = (MODE == MODE_H) ? 32 : (MODE == MODE_TQKV) ? 48
                 : (MODE == MODE_GQKV) ? 96 : 64;
    const bool OUTH = (MODE == MODE_H || MODE == MODE_XG || MODE == MODE_GQKV);

    __shared__ __align__(16) h16 As[128 * ASTRIDE];
    __shared__ __align__(16) h16 BsT[32 * BSTRIDE];   // [col][k]
    int tid = threadIdx.x;
    int wave = tid >> 5, lane = tid & 31;
    int row0 = blockIdx.x * 128;
    int col0 = blockIdx.y * 32;
    int halfsel = lane >> 4, m = lane & 15;
    v8h zero = {};

    v8f acc[2] = {};
#pragma unroll
    for (int k0 = 0; k0 < K; k0 += 32) {
        // stage A: 512 chunks of 8 halfs, 2 per thread, branchless clamp
#pragma unroll
        for (int c = 0; c < 2; ++c) {
            int chunk = tid + 256 * c;
            int rr = chunk >> 2, koff = (chunk & 3) * 8;
            int r = row0 + rr;
            int kg = k0 + koff;
            int rc = r < M ? r : M - 1;
            int kc = kg < K ? kg : 0;
            v8h v = *(const v8h*)(A + (long)rc * K + kc);
            if (r >= M || kg >= K) v = zero;
            *(v8h*)(As + rr * ASTRIDE + koff) = v;
        }
        // stage B: 128 chunks (32 cols x 4 chunks)
        if (tid < 128) {
            int nn = tid >> 2, koff = (tid & 3) * 8;
            int col = col0 + nn;
            int kg = k0 + koff;
            int cc = col < NC ? col : NC - 1;
            int kc = kg < K ? kg : 0;
            v8h v = *(const v8h*)(Bwt + (long)cc * K + kc);
            if (col >= NC || kg >= K) v = zero;
            *(v8h*)(BsT + nn * BSTRIDE + koff) = v;
        }
        __syncthreads();
        v16h af  = load_a_frag(As + (wave * 16 + m) * ASTRIDE, halfsel);
        v16h bf0 = load_b_fragT(BsT + m * BSTRIDE, halfsel);
        v16h bf1 = load_b_fragT(BsT + (16 + m) * BSTRIDE, halfsel);
        acc[0] = __builtin_amdgcn_wmma_f32_16x16x32_f16(false, af, false, bf0,
                                                        (short)0, acc[0], false, false);
        acc[1] = __builtin_amdgcn_wmma_f32_16x16x32_f16(false, af, false, bf1,
                                                        (short)0, acc[1], false, false);
        __syncthreads();
    }
#pragma unroll
    for (int tdx = 0; tdx < 2; ++tdx) {
        int col = col0 + tdx * 16 + m;
        if (col < NC) {
            float bv = bias ? bias[col] : 0.f;
#pragma unroll
            for (int r = 0; r < 8; ++r) {
                int rowm = row0 + wave * 16 + r + 8 * halfsel;
                if (rowm < M) {
                    long co = c_off_row<MODE>(rowm);
                    float v = acc[tdx][r] + bv;
                    if (OUTH) ((h16*)Cout)[co + col] = (h16)v;
                    else      ((float*)Cout)[co + col] = v;
                }
            }
        }
    }
}

// ---------------------------------------------------------------------------
// GCN aggregation (per-batch WMMA GEMM): dst[w, t, c+dst_off] =
//   sum_v adp16[b,w,v] * hcat[b, v, t, c+src_off]  (M=325, N=384, K=325->336)
// blockIdx.y == t (each 32-col tile is exactly one timestep).
// ---------------------------------------------------------------------------
__global__ __launch_bounds__(256) void k_gcn_agg(
    const h16* __restrict__ adp16, h16* __restrict__ hcat,
    int src_off, int dst_off)
{
    __shared__ __align__(16) h16 As[128 * ASTRIDE];
    __shared__ __align__(16) h16 BsT[32 * BSTRIDE];
    int b = blockIdx.z;
    int row0 = blockIdx.x * 128;
    int t = blockIdx.y;                 // 0..11; cols j = t*32 + c
    int tid = threadIdx.x;
    int wave = tid >> 5, lane = tid & 31;
    int halfsel = lane >> 4, m = lane & 15;
    v8h zero = {};

    const h16* adpb = adp16 + (long)b * NN * ANP;
    const h16* hb   = hcat  + (long)b * NN * TT * 96;

    v8f acc[2] = {};
    for (int k0 = 0; k0 < ANP; k0 += 32) {
        // stage A from padded adp rows (linear, chunked)
#pragma unroll
        for (int c = 0; c < 2; ++c) {
            int chunk = tid + 256 * c;
            int rr = chunk >> 2, koff = (chunk & 3) * 8;
            int w = row0 + rr;
            int kg = k0 + koff;
            int wc = w < NN ? w : NN - 1;
            int kc = kg < ANP ? kg : 0;
            v8h v = *(const v8h*)(adpb + (long)wc * ANP + kc);
            if (w >= NN || kg >= ANP) v = zero;
            *(v8h*)(As + rr * ASTRIDE + koff) = v;
        }
        // stage B (transpose of 32x32 contiguous panel at [v][t*96+src_off])
        if (tid < 128) {
            int vl = tid >> 2, coff = (tid & 3) * 8;
            int v = k0 + vl;
            int vc = v < NN ? v : NN - 1;
            v8h val = *(const v8h*)(hb + (long)vc * (TT * 96) + t * 96 + src_off + coff);
            if (v >= NN) val = zero;
#pragma unroll
            for (int i = 0; i < 8; ++i)
                BsT[(coff + i) * BSTRIDE + vl] = val[i];
        }
        __syncthreads();
        v16h af  = load_a_frag(As + (wave * 16 + m) * ASTRIDE, halfsel);
        v16h bf0 = load_b_fragT(BsT + m * BSTRIDE, halfsel);
        v16h bf1 = load_b_fragT(BsT + (16 + m) * BSTRIDE, halfsel);
        acc[0] = __builtin_amdgcn_wmma_f32_16x16x32_f16(false, af, false, bf0,
                                                        (short)0, acc[0], false, false);
        acc[1] = __builtin_amdgcn_wmma_f32_16x16x32_f16(false, af, false, bf1,
                                                        (short)0, acc[1], false, false);
        __syncthreads();
    }
#pragma unroll
    for (int tdx = 0; tdx < 2; ++tdx) {
        int c = tdx * 16 + m;
#pragma unroll
        for (int r = 0; r < 8; ++r) {
            int w = row0 + wave * 16 + r + 8 * halfsel;
            if (w < NN)
                hcat[(((long)b * NN + w) * TT + t) * 96 + dst_off + c] = (h16)acc[tdx][r];
        }
    }
}

// ---------------------------------------------------------------------------
// Temporal attention (scalar, tiny: T=12, hd=8). Writes att48 cols 0..15.
// ---------------------------------------------------------------------------
__global__ __launch_bounds__(32) void k_temporal_attn(
    const float* __restrict__ tqkv, h16* __restrict__ att48)
{
    int bn = blockIdx.x;
    int b = bn / NN, n = bn % NN;
    int lane = threadIdx.x;
    __shared__ float sh[TT * 48];
    for (int idx = lane; idx < TT * 48; idx += 32) {
        int t = idx / 48, c = idx % 48;
        sh[idx] = tqkv[(((long)b * TT + t) * NN + n) * 48 + c];
    }
    __syncthreads();
    if (lane < TT) {
        int t = lane;
        for (int h = 0; h < 2; ++h) {
            float q[8];
#pragma unroll
            for (int d = 0; d < 8; ++d) q[d] = sh[t * 48 + h * 8 + d];
            float s[TT];
            float mx = -1e30f;
#pragma unroll
            for (int k = 0; k < TT; ++k) {
                float dot = 0.f;
#pragma unroll
                for (int d = 0; d < 8; ++d) dot += q[d] * sh[k * 48 + 16 + h * 8 + d];
                s[k] = dot * SCALE_HD;
                mx = fmaxf(mx, s[k]);
            }
            float sum = 0.f;
#pragma unroll
            for (int k = 0; k < TT; ++k) { s[k] = __expf(s[k] - mx); sum += s[k]; }
            float inv = 1.f / sum;
            float o[8] = {0.f, 0.f, 0.f, 0.f, 0.f, 0.f, 0.f, 0.f};
#pragma unroll
            for (int k = 0; k < TT; ++k)
#pragma unroll
                for (int d = 0; d < 8; ++d) o[d] += s[k] * sh[k * 48 + 32 + h * 8 + d];
#pragma unroll
            for (int d = 0; d < 8; ++d)
                att48[(((long)b * TT + t) * NN + n) * 48 + h * 8 + d] = (h16)(o[d] * inv);
        }
    }
}

// ---------------------------------------------------------------------------
// Geo attention: one wave per (qstrip of 16, b*T, head). WMMA QK^T (hd=8
// zero-padded to K=32) and PV (K=325 in 32-steps). Writes att48 cols 16..47.
// ---------------------------------------------------------------------------
__global__ __launch_bounds__(32) void k_geo_attn(
    const h16* __restrict__ gqkv, const unsigned char* __restrict__ geo_mask,
    h16* __restrict__ att48)
{
    int qs = blockIdx.x;            // 21 strips of 16 query rows
    int bt = blockIdx.y;            // b*T + t
    int h  = blockIdx.z;            // 4 heads
    int lane = threadIdx.x;
    int halfsel = lane >> 4, m = lane & 15;

    __shared__ __align__(16) h16   Qs[16 * 32];
    __shared__ __align__(16) h16   KTs[16 * BSTRIDE];   // [node][dpad]
    __shared__ __align__(16) h16   VTs[16 * BSTRIDE];   // [d][node]
    __shared__ __align__(16) float S[16 * 336];
    __shared__ __align__(16) h16   P[16 * 336];
    __shared__ float rowinv[16];

    const h16* base = gqkv + (long)bt * NN * 96;

    for (int idx = lane; idx < 512; idx += 32) {
        int r = idx >> 5, kk = idx & 31;
        int qn = qs * 16 + r;
        int qc = qn < NN ? qn : NN - 1;
        int kc = kk < 8 ? kk : 7;
        h16 v = base[(long)qc * 96 + h * 8 + kc];
        if (qn >= NN || kk >= 8) v = (h16)0.f;
        Qs[idx] = v;
    }
    __syncthreads();
    v16h af = load_a_frag(Qs + m * 32, halfsel);

    for (int kt = 0; kt < 21; ++kt) {
        for (int idx = lane; idx < 512; idx += 32) {
            int nn = idx >> 5, dd = idx & 31;
            int node = kt * 16 + nn;
            int nc = node < NN ? node : NN - 1;
            int dc = dd < 8 ? dd : 7;
            h16 v = base[(long)nc * 96 + 32 + h * 8 + dc];
            if (node >= NN || dd >= 8) v = (h16)0.f;
            KTs[nn * BSTRIDE + dd] = v;
        }
        __syncthreads();
        v16h bf = load_b_fragT(KTs + m * BSTRIDE, halfsel);
        v8f c = {};
        c = __builtin_amdgcn_wmma_f32_16x16x32_f16(false, af, false, bf,
                                                   (short)0, c, false, false);
        int node = kt * 16 + m;
#pragma unroll
        for (int r = 0; r < 8; ++r) {
            int row = r + 8 * halfsel;
            int qn = qs * 16 + row;
            int qc = qn < NN ? qn : NN - 1;
            int nc = node < NN ? node : NN - 1;
            unsigned char mk = geo_mask[(long)qc * NN + nc];
            float v = c[r] * SCALE_HD;
            if (node >= NN || qn >= NN || mk) v = -1e30f;
            S[row * 336 + node] = v;
        }
        __syncthreads();
    }

    if (lane < 16) {
        float mx = -1e30f;
        for (int j = 0; j < NN; ++j) mx = fmaxf(mx, S[lane * 336 + j]);
        float sum = 0.f;
        for (int j = 0; j < 336; ++j) {
            float e = (j < NN) ? __expf(S[lane * 336 + j] - mx) : 0.f;
            sum += e;
            P[lane * 336 + j] = (h16)e;
        }
        rowinv[lane] = 1.f / sum;
    }
    __syncthreads();

    v8f oacc = {};
    for (int k0 = 0; k0 < 336; k0 += 32) {
        for (int idx = lane; idx < 512; idx += 32) {
            int nl = idx >> 4, dd = idx & 15;
            int node = k0 + nl;
            int nc = node < NN ? node : NN - 1;
            int dc = dd < 8 ? dd : 7;
            h16 v = base[(long)nc * 96 + 64 + h * 8 + dc];
            if (node >= NN || dd >= 8) v = (h16)0.f;
            VTs[dd * BSTRIDE + nl] = v;
        }
        __syncthreads();
        v16h pf = load_a_frag(P + m * 336 + k0, halfsel);
        v16h vf = load_b_fragT(VTs + m * BSTRIDE, halfsel);
        oacc = __builtin_amdgcn_wmma_f32_16x16x32_f16(false, pf, false, vf,
                                                      (short)0, oacc, false, false);
        __syncthreads();
    }
    if (m < 8) {
#pragma unroll
        for (int r = 0; r < 8; ++r) {
            int row = r + 8 * halfsel;
            int qn = qs * 16 + row;
            if (qn < NN)
                att48[((long)bt * NN + qn) * 48 + 16 + h * 8 + m] =
                    (h16)(oacc[r] * rowinv[row]);
        }
    }
}

// ---------------------------------------------------------------------------
// host launch
// ---------------------------------------------------------------------------
extern "C" void kernel_launch(void* const* d_in, const int* in_sizes, int n_in,
                              void* d_out, int out_size, void* d_ws, size_t ws_size,
                              hipStream_t stream) {
    (void)in_sizes; (void)n_in; (void)out_size; (void)ws_size;
    const float* x       = (const float*)d_in[0];
    const int*   ind     = (const int*)d_in[1];
    const unsigned char* geo_mask = (const unsigned char*)d_in[4]; // bool array
    const float* p1      = (const float*)d_in[5];
    const float* p2      = (const float*)d_in[6];
    const float* p3      = (const float*)d_in[7];
    const float* pk      = (const float*)d_in[8];
    const float* W_r1    = (const float*)d_in[9];
    const float* b_r1    = (const float*)d_in[10];
    const float* W_gcn   = (const float*)d_in[11];
    const float* b_gcn   = (const float*)d_in[12];
    const float* W_r2    = (const float*)d_in[13];
    const float* b_r2    = (const float*)d_in[14];
    const float* Wtq     = (const float*)d_in[15];
    const float* Wtk     = (const float*)d_in[16];
    const float* Wtv     = (const float*)d_in[17];
    const float* Wgq     = (const float*)d_in[18];
    const float* Wgk     = (const float*)d_in[19];
    const float* Wgv     = (const float*)d_in[20];
    const float* W_proj  = (const float*)d_in[21];
    const float* b_proj  = (const float*)d_in[22];
    float* out = (float*)d_out;

    char* ws = (char*)d_ws;
    size_t off = 0;
    auto alloc = [&](size_t bytes) -> void* {
        void* p = ws + off;
        off = (off + bytes + 255) & ~(size_t)255;
        return p;
    };
    float* tmp1    = (float*)alloc((size_t)BB * 1600 * sizeof(float));
    h16*   adp16   = (h16*)  alloc((size_t)BB * NN * ANP * sizeof(h16));
    h16*   x16     = (h16*)  alloc((size_t)MROW * 64 * sizeof(h16));
    h16*   hcat    = (h16*)  alloc((size_t)BB * NN * TT * 96 * sizeof(h16));
    h16*   xg16    = (h16*)  alloc((size_t)MROW * 64 * sizeof(h16));
    float* tqkv    = (float*)alloc((size_t)MROW * 48 * sizeof(float));
    h16*   gqkv    = (h16*)  alloc((size_t)MROW * 96 * sizeof(h16));
    h16*   att48   = (h16*)  alloc((size_t)MROW * 48 * sizeof(h16));
    h16*   Bw1     = (h16*)  alloc((size_t)32 * 64 * sizeof(h16));
    h16*   Bcomb   = (h16*)  alloc((size_t)64 * 96 * sizeof(h16));
    float* biascomb= (float*)alloc((size_t)64 * sizeof(float));
    h16*   Bt      = (h16*)  alloc((size_t)48 * 64 * sizeof(h16));
    h16*   Bg      = (h16*)  alloc((size_t)96 * 64 * sizeof(h16));
    h16*   Bproj   = (h16*)  alloc((size_t)64 * 48 * sizeof(h16));

    const int GX = (MROW + 127) / 128;  // 488

    k_f32_to_f16<<<2048, 256, 0, stream>>>(x, x16, (long)MROW * 64);
    k_prep_weights<<<32, 256, 0, stream>>>(W_r1, W_gcn, b_gcn, W_r2, b_r2,
                                           Wtq, Wtk, Wtv, Wgq, Wgk, Wgv, W_proj,
                                           Bw1, Bcomb, biascomb, Bt, Bg, Bproj);
    k_tmp1<<<BB, 256, 0, stream>>>(p1, ind, pk, tmp1);
    k_adp<<<BB * NN, 128, 0, stream>>>(p2, p3, tmp1, adp16);
    // h = x @ W_r1^T + b_r1 -> hcat ch 0..31
    k_row_gemm<MODE_H><<<dim3(GX, 1), 256, 0, stream>>>(x16, Bw1, b_r1, hcat, MROW);
    // GCN order-2: x1 = adp@h (ch 32..63), x2 = adp@x1 (ch 64..95)
    k_gcn_agg<<<dim3(3, TT, BB), 256, 0, stream>>>(adp16, hcat, 0, 32);
    k_gcn_agg<<<dim3(3, TT, BB), 256, 0, stream>>>(adp16, hcat, 32, 64);
    // xg = hcat @ (W_r2@W_gcn)^T + biascomb
    k_row_gemm<MODE_XG><<<dim3(GX, 2), 256, 0, stream>>>(hcat, Bcomb, biascomb, xg16, MROW);
    // fused temporal QKV (f32 out)
    k_row_gemm<MODE_TQKV><<<dim3(GX, 2), 256, 0, stream>>>(xg16, Bt, nullptr, tqkv, MROW);
    // fused geo QKV (f16 out)
    k_row_gemm<MODE_GQKV><<<dim3(GX, 3), 256, 0, stream>>>(xg16, Bg, nullptr, gqkv, MROW);
    // temporal attention -> att48 cols 0..15
    k_temporal_attn<<<BB * NN, 32, 0, stream>>>(tqkv, att48);
    // geo attention -> att48 cols 16..47
    k_geo_attn<<<dim3(21, BB * TT, 4), 32, 0, stream>>>(gqkv, geo_mask, att48);
    // output projection (f32 out)
    k_row_gemm<MODE_PROJ><<<dim3(GX, 2), 256, 0, stream>>>(att48, Bproj, b_proj, out, MROW);
}
